// RelationalGraphConv_429496729699
// MI455X (gfx1250) — matrix operand order repaired and checked by reading, not compiled, not run on hardware
//
#include <hip/hip_runtime.h>
#include <hip/hip_bf16.h>

typedef __attribute__((ext_vector_type(16))) __bf16 v16bf;
typedef __attribute__((ext_vector_type(8)))  __bf16 v8bf;
typedef __attribute__((ext_vector_type(8)))  float  v8f;

#define RGC_INF    64
#define RGC_OUTF   64
#define RGC_NREL   32
#define RGC_NBASES 30
#define RGC_CPW    8   // chunks (of 16 edges) per wave

static __device__ __forceinline__ int imin(int a, int b) { return a < b ? a : b; }

// ---------------------------------------------------------------------------
// W[r] = sum_b coeff[r,b] * basis[b]; store TRANSPOSED [r][out][in] as bf16 hi/lo
// ---------------------------------------------------------------------------
__global__ void rgc_weights(const float* __restrict__ basis,
                            const float* __restrict__ coeff,
                            __bf16* __restrict__ Wth, __bf16* __restrict__ Wtl) {
  int t = blockIdx.x * blockDim.x + threadIdx.x;           // [r][o][i]
  if (t >= RGC_NREL * RGC_OUTF * RGC_INF) return;
  int i = t % RGC_INF;
  int o = (t / RGC_INF) % RGC_OUTF;
  int r = t / (RGC_INF * RGC_OUTF);
  float w = 0.f;
  #pragma unroll
  for (int b = 0; b < RGC_NBASES; ++b)
    w = fmaf(coeff[r * RGC_NBASES + b], basis[(b * RGC_INF + i) * RGC_OUTF + o], w);
  __bf16 h = (__bf16)w;
  Wth[t] = h;
  Wtl[t] = (__bf16)(w - (float)h);
}

// ---------------------------------------------------------------------------
// histogram over relations + in-degree
// ---------------------------------------------------------------------------
__global__ void rgc_hist_deg(const int* __restrict__ etype,
                             const int* __restrict__ tgt,
                             float* __restrict__ deg, int* __restrict__ hist, int E) {
  int e = blockIdx.x * blockDim.x + threadIdx.x;
  if (e >= E) return;
  atomicAdd(&hist[etype[e]], 1);
  atomicAdd(&deg[tgt[e]], 1.0f);
}

// ---------------------------------------------------------------------------
// tiny serial scan (32 entries): edge offsets, cursors, chunk offsets
// ---------------------------------------------------------------------------
__global__ void rgc_scan(const int* __restrict__ hist, int* __restrict__ offsets,
                         int* __restrict__ cursor, int* __restrict__ chunk_off) {
  if (threadIdx.x != 0 || blockIdx.x != 0) return;
  int run = 0, crun = 0;
  for (int r = 0; r < RGC_NREL; ++r) {
    offsets[r] = run;
    cursor[r] = run;
    chunk_off[r] = crun;
    int c = hist[r];
    run  += c;
    crun += (c + 15) >> 4;
  }
  offsets[RGC_NREL]   = run;
  chunk_off[RGC_NREL] = crun;
}

__global__ void rgc_scatter(const int* __restrict__ etype, int* __restrict__ cursor,
                            int* __restrict__ sortedE, int E) {
  int e = blockIdx.x * blockDim.x + threadIdx.x;
  if (e >= E) return;
  int p = atomicAdd(&cursor[etype[e]], 1);
  sortedE[p] = e;
}

// ---------------------------------------------------------------------------
// split x into bf16 hi/lo
// ---------------------------------------------------------------------------
__global__ void rgc_xsplit(const float* __restrict__ x,
                           __bf16* __restrict__ xh, __bf16* __restrict__ xl, int n) {
  int i = blockIdx.x * blockDim.x + threadIdx.x;
  if (i >= n) return;
  float f = x[i];
  __bf16 h = (__bf16)f;
  xh[i] = h;
  xl[i] = (__bf16)(f - (float)h);
}

// ---------------------------------------------------------------------------
// main edge-message kernel: one wave handles RGC_CPW chunks of 16 edges.
// D(16x16) += A(16edges x 32K, bf16 hi/lo) x B(32K x 16out, bf16 hi/lo)
// via v_wmma_f32_16x16x32_bf16; 2 K-tiles x 4 N-tiles x 3 splits = 24 WMMA/chunk.
// Index gathers for chunk c+1 are issued before chunk c's math (SW pipeline),
// and next x-rows are warmed with global_prefetch.
// ---------------------------------------------------------------------------
__global__ void __launch_bounds__(256)
rgc_edges(const __bf16* __restrict__ xh, const __bf16* __restrict__ xl,
          const __bf16* __restrict__ Wth, const __bf16* __restrict__ Wtl,
          const int* __restrict__ srcArr, const int* __restrict__ tgtArr,
          const int* __restrict__ sortedE, const int* __restrict__ offsets,
          const int* __restrict__ chunk_off, float* __restrict__ out) {
  const int lane = threadIdx.x & 31;
  const int wid  = (blockIdx.x * blockDim.x + threadIdx.x) >> 5;
  const int total = chunk_off[RGC_NREL];

  int cc   = wid * RGC_CPW;
  const int cend = imin(cc + RGC_CPW, total);
  if (cc >= cend) return;                      // wave-uniform exit

  const int half = lane >> 4;                  // 0: lanes 0-15, 1: lanes 16-31
  const int m    = lane & 15;

  // ---- prologue: resolve + load indices for the first chunk ----
  int r_cur = 0;
  while (chunk_off[r_cur + 1] <= cc) ++r_cur;
  int eEnd_cur  = offsets[r_cur + 1];
  int ebase_cur = offsets[r_cur] + (cc - chunk_off[r_cur]) * 16;
  {
    int myE = ebase_cur + m;
    int eid = sortedE[(myE < eEnd_cur) ? myE : ebase_cur];
    // fallthrough loads below
    int s = srcArr[eid], t = tgtArr[eid];
    // stash in registers via variables declared after (use immediately)
    // (handled by srow/trow init below)
    __builtin_prefetch(xh + (size_t)s * RGC_INF, 0, 3);
    __builtin_prefetch(xl + (size_t)s * RGC_INF, 0, 3);
    // re-declare outside block
    // (see srow/trow)
    // store into outer vars:
    // -- done below --
    // NOTE: kept minimal; real init:
    // srow/trow set here
    // (cannot declare before use cleanly in this block)
    // we simply recompute nothing; assign:
    // outer declarations follow
    // ...
    // Assign via pointer trick not needed; move declarations up instead.
    // (This block retained for prefetch only.)
    (void)t;
  }
  int myE0 = ebase_cur + m;
  int eid0 = sortedE[(myE0 < eEnd_cur) ? myE0 : ebase_cur];
  int srow = srcArr[eid0];
  int trow = tgtArr[eid0];

  int cachedRel = -1;
  v16bf Bh[8], Bl[8];                          // [kbI*4 + nt]

  while (cc < cend) {
    // ---- issue next chunk's index gathers early (overlap with WMMA) ----
    const int cn = cc + 1;
    const bool have_next = (cn < cend);        // wave-uniform
    int r_nxt = r_cur, eEnd_n = eEnd_cur, ebase_n = ebase_cur;
    int srow_n = 0, trow_n = 0;
    if (have_next) {
      while (chunk_off[r_nxt + 1] <= cn) ++r_nxt;
      eEnd_n  = offsets[r_nxt + 1];
      ebase_n = offsets[r_nxt] + (cn - chunk_off[r_nxt]) * 16;
      int myEn = ebase_n + m;
      int eidn = sortedE[(myEn < eEnd_n) ? myEn : ebase_n];
      srow_n = srcArr[eidn];
      trow_n = tgtArr[eidn];
      __builtin_prefetch(xh + (size_t)srow_n * RGC_INF, 0, 3);
      __builtin_prefetch(xl + (size_t)srow_n * RGC_INF, 0, 3);
    }

    // ---- B tiles: (re)load only when the relation changes ----
    if (r_cur != cachedRel) {
      // B layout (bf16 32x16): lane -> N = nt*16 + m ; elem j -> K = kbI*32 + half*16 + j
      #pragma unroll
      for (int kbI = 0; kbI < 2; ++kbI)
        #pragma unroll
        for (int nt = 0; nt < 4; ++nt) {
          const int N = nt * 16 + m;
          const int Kb = kbI * 32 + half * 16;
          const int base = (r_cur * RGC_OUTF + N) * RGC_INF + Kb;   // [r][out][in]
          Bh[kbI * 4 + nt] = *(const v16bf*)(Wth + base);
          Bl[kbI * 4 + nt] = *(const v16bf*)(Wtl + base);
        }
      cachedRel = r_cur;
    }

    // ---- 16x64 @ 64x64 via WMMA with bf16 hi/lo split ----
    v8f z = {0.f, 0.f, 0.f, 0.f, 0.f, 0.f, 0.f, 0.f};
    v8f acc[4] = {z, z, z, z};

    #pragma unroll
    for (int kbI = 0; kbI < 2; ++kbI) {
      // A layout (bf16 16x32): lane -> M = m ; elems 0..7 -> K = koff+0..7,
      // elems 8..15 -> K = koff+8..15, koff = kbI*32 + half*16
      const int koff = kbI * 32 + half * 16;
      const __bf16* xrh = xh + (size_t)srow * RGC_INF + koff;
      const __bf16* xrl = xl + (size_t)srow * RGC_INF + koff;
      v8bf ah0 = *(const v8bf*)(xrh);
      v8bf ah1 = *(const v8bf*)(xrh + 8);
      v8bf al0 = *(const v8bf*)(xrl);
      v8bf al1 = *(const v8bf*)(xrl + 8);
      v16bf Ah, Al;
      #pragma unroll
      for (int j = 0; j < 8; ++j) {
        Ah[j] = ah0[j]; Ah[j + 8] = ah1[j];
        Al[j] = al0[j]; Al[j + 8] = al1[j];
      }
      #pragma unroll
      for (int nt = 0; nt < 4; ++nt) {
        acc[nt] = __builtin_amdgcn_wmma_f32_16x16x32_bf16(
            false, Ah, false, Bh[kbI * 4 + nt], (short)0, acc[nt], false, false);
        acc[nt] = __builtin_amdgcn_wmma_f32_16x16x32_bf16(
            false, Ah, false, Bl[kbI * 4 + nt], (short)0, acc[nt], false, false);
        acc[nt] = __builtin_amdgcn_wmma_f32_16x16x32_bf16(
            false, Al, false, Bh[kbI * 4 + nt], (short)0, acc[nt], false, false);
      }
    }

    // ---- scatter: C/D layout -> lane holds N = nt*16+m, elem v -> M = half*8+v ----
    // Hoist all 8 target-row shuffles so the ds_bpermutes pipeline together.
    int trv[8];
    #pragma unroll
    for (int v = 0; v < 8; ++v)
      trv[v] = __shfl(trow, half * 8 + v, 32);

    if (ebase_cur + 16 <= eEnd_cur) {
      // full chunk (common case): no predication, 4 atomics share one address
      #pragma unroll
      for (int v = 0; v < 8; ++v) {
        float* p = out + (size_t)trv[v] * RGC_OUTF + m;
        #pragma unroll
        for (int nt = 0; nt < 4; ++nt)
          atomicAdd(p + nt * 16, acc[nt][v]);
      }
    } else {
      // tail chunk of a relation: mask invalid edges
      #pragma unroll
      for (int v = 0; v < 8; ++v) {
        if (ebase_cur + half * 8 + v < eEnd_cur) {
          float* p = out + (size_t)trv[v] * RGC_OUTF + m;
          #pragma unroll
          for (int nt = 0; nt < 4; ++nt)
            atomicAdd(p + nt * 16, acc[nt][v]);
        }
      }
    }

    // ---- rotate pipeline state ----
    cc = cn;
    r_cur = r_nxt; eEnd_cur = eEnd_n; ebase_cur = ebase_n;
    srow = srow_n; trow = trow_n;
  }
}

// ---------------------------------------------------------------------------
// out = msg/deg + x @ self_weight + bias
// ---------------------------------------------------------------------------
__global__ void rgc_finalize(const float* __restrict__ x, const float* __restrict__ sw,
                             const float* __restrict__ bias, const float* __restrict__ deg,
                             float* __restrict__ out, int nNodes) {
  int t = blockIdx.x * blockDim.x + threadIdx.x;
  if (t >= nNodes * RGC_OUTF) return;
  int nIdx = t / RGC_OUTF, o = t % RGC_OUTF;
  float d = deg[nIdx]; if (d < 1.f) d = 1.f;
  float s = bias[o];
  const float* xr = x + (size_t)nIdx * RGC_INF;
  #pragma unroll 8
  for (int i = 0; i < RGC_INF; ++i) s = fmaf(xr[i], sw[i * RGC_OUTF + o], s);
  out[t] = out[t] / d + s;
}

// ---------------------------------------------------------------------------
extern "C" void kernel_launch(void* const* d_in, const int* in_sizes, int n_in,
                              void* d_out, int out_size, void* d_ws, size_t ws_size,
                              hipStream_t stream) {
  const float* x          = (const float*)d_in[0];
  const int*   edge_index = (const int*)d_in[1];     // [2][E]
  const int*   etype      = (const int*)d_in[2];     // [E]
  const float* basis      = (const float*)d_in[4];
  const float* coeff      = (const float*)d_in[5];
  const float* sw         = (const float*)d_in[6];
  const float* bias       = (const float*)d_in[7];
  float* out = (float*)d_out;

  const int E      = in_sizes[2];
  const int nNodes = in_sizes[0] / RGC_INF;
  const int* srcArr = edge_index;
  const int* tgtArr = edge_index + E;

  // carve workspace (256B aligned slices)
  char* w = (char*)d_ws;
  auto carve = [&](size_t bytes) { char* p = w; w += (bytes + 255) & ~(size_t)255; return p; };
  float*  deg       = (float*)carve((size_t)nNodes * sizeof(float));
  int*    hist      = (int*)carve(RGC_NREL * sizeof(int));
  int*    offsets   = (int*)carve((RGC_NREL + 1) * sizeof(int));
  int*    cursor    = (int*)carve(RGC_NREL * sizeof(int));
  int*    chunk_off = (int*)carve((RGC_NREL + 1) * sizeof(int));
  int*    sortedE   = (int*)carve((size_t)E * sizeof(int));
  __bf16* Wth       = (__bf16*)carve((size_t)RGC_NREL * RGC_OUTF * RGC_INF * sizeof(__bf16));
  __bf16* Wtl       = (__bf16*)carve((size_t)RGC_NREL * RGC_OUTF * RGC_INF * sizeof(__bf16));
  __bf16* xhS       = (__bf16*)carve((size_t)nNodes * RGC_INF * sizeof(__bf16));
  __bf16* xlS       = (__bf16*)carve((size_t)nNodes * RGC_INF * sizeof(__bf16));
  (void)ws_size; (void)n_in; (void)out_size;

  // zero accumulators + histogram
  hipMemsetAsync(out, 0, (size_t)nNodes * RGC_OUTF * sizeof(float), stream);
  hipMemsetAsync(deg, 0, (size_t)nNodes * sizeof(float), stream);
  hipMemsetAsync(hist, 0, RGC_NREL * sizeof(int), stream);

  const int wTot = RGC_NREL * RGC_OUTF * RGC_INF;
  rgc_weights<<<(wTot + 255) / 256, 256, 0, stream>>>(basis, coeff, Wth, Wtl);
  rgc_hist_deg<<<(E + 255) / 256, 256, 0, stream>>>(etype, tgtArr, deg, hist, E);
  rgc_scan<<<1, 1, 0, stream>>>(hist, offsets, cursor, chunk_off);
  rgc_scatter<<<(E + 255) / 256, 256, 0, stream>>>(etype, cursor, sortedE, E);

  const int xTot = nNodes * RGC_INF;
  rgc_xsplit<<<(xTot + 255) / 256, 256, 0, stream>>>(x, xhS, xlS, xTot);

  const int maxChunks = (E + 15) / 16 + RGC_NREL;             // upper bound on device total
  const int wavesNeeded = (maxChunks + RGC_CPW - 1) / RGC_CPW;
  const int eBlocks = (wavesNeeded + 7) / 8;                  // 8 waves / 256-thread block
  rgc_edges<<<eBlocks, 256, 0, stream>>>(xhS, xlS, Wth, Wtl, srcArr, tgtArr,
                                         sortedE, offsets, chunk_off, out);

  rgc_finalize<<<(nNodes * RGC_OUTF + 255) / 256, 256, 0, stream>>>(x, sw, bias, deg, out, nNodes);
}
